// LIIF_29076928594428
// MI455X (gfx1250) — matrix operand order repaired and checked by reading, not compiled, not run on hardware
//
#include <hip/hip_runtime.h>
#include <hip/hip_bf16.h>
#include <cstdint>
#include <cstddef>

typedef __attribute__((ext_vector_type(16))) _Float16 v16h;
typedef __attribute__((ext_vector_type(8)))  _Float16 v8h;
typedef __attribute__((ext_vector_type(8)))  float    v8f;
typedef int v4i __attribute__((vector_size(16)));

#if __has_builtin(__builtin_amdgcn_global_load_async_to_lds_b128)
#define HAS_ASYNC_LDS 1
typedef __attribute__((address_space(1))) v4i g_v4i;
typedef __attribute__((address_space(3))) v4i l_v4i;
#else
#define HAS_ASYNC_LDS 0
#endif

__device__ __forceinline__ void sched_fence() {
#if __has_builtin(__builtin_amdgcn_sched_barrier)
  __builtin_amdgcn_sched_barrier(0);
#endif
}

namespace {
constexpr int Bn   = 2;
constexpr int Qn   = 2048;
constexpr int BQ   = Bn * Qn;        // 4096
constexpr int NOFF = 36;
constexpr int CH   = 6;              // offsets per chunk
constexpr int NCH  = NOFF / CH;      // 6 chunks
constexpr int K0   = 578;
constexpr int KP0  = 608;            // padded to multiple of 32
constexpr int NMID = 256;
constexpr int NOUT = 27;
constexpr int NP4  = 32;             // padded output width
constexpr int ROWS = CH * BQ;        // 24576 rows per chunk
constexpr float INV64 = 1.0f / 64.0f;
constexpr float LOG1E4_D16 = 0.57564627324851143f; // ln(10000)/16
static_assert((ROWS / 16) % 8 == 0, "every wave active -> barriers safe");
}

__device__ __forceinline__ float offv(int t) {
  // offsets {-3,-2,-1,1,2,3}
  return (float)(t < 3 ? t - 3 : t - 2);
}

// Shared helper: nearest-grid-cell lookup + its center coordinate.
__device__ __forceinline__ void grid_pos(const float* cd, float vx, float vy,
                                         int& iy, int& ix, float& qc0, float& qc1) {
  float cc0 = cd[8] + vx * INV64 + 1e-6f;
  float cc1 = cd[9] + vy * INV64 + 1e-6f;
  cc0 = fminf(fmaxf(cc0, -1.f + 1e-6f), 1.f - 1e-6f);
  cc1 = fminf(fmaxf(cc1, -1.f + 1e-6f), 1.f - 1e-6f);
  float fy = rintf(((cc0 + 1.f) * 64.f - 1.f) * 0.5f);   // round-to-even == jnp.round
  float fx = rintf(((cc1 + 1.f) * 64.f - 1.f) * 0.5f);
  iy = (int)fminf(fmaxf(fy, 0.f), 63.f);
  ix = (int)fminf(fmaxf(fx, 0.f), 63.f);
  qc0 = -1.f + (2.f * (float)iy + 1.f) * INV64;
  qc1 = -1.f + (2.f * (float)ix + 1.f) * INV64;
}

// ---------- 3x3 conv encoder (tiny, scalar f32) ----------
__global__ void encoder_kernel(const float* __restrict__ inp,
                               const float* __restrict__ ew,
                               const float* __restrict__ eb,
                               float* __restrict__ feat) {
  int idx = blockIdx.x * blockDim.x + threadIdx.x;
  if (idx >= Bn * 64 * 64 * 64) return;
  int x = idx & 63, y = (idx >> 6) & 63, c = (idx >> 12) & 63, b = idx >> 18;
  float acc = eb[c];
#pragma unroll
  for (int ci = 0; ci < 3; ++ci)
#pragma unroll
    for (int ky = 0; ky < 3; ++ky)
#pragma unroll
      for (int kx = 0; kx < 3; ++kx) {
        int yy = y + ky - 1, xx = x + kx - 1;
        if ((unsigned)yy < 64u && (unsigned)xx < 64u)
          acc += inp[(((size_t)b * 3 + ci) * 64 + yy) * 64 + xx] *
                 ew[((c * 3 + ci) * 3 + ky) * 3 + kx];
      }
  feat[idx] = acc;
}

// ---------- f32 [K][N] weights -> f16 transposed [Npad][Kpad] ----------
__global__ void cvt_w_kernel(const float* __restrict__ w, _Float16* __restrict__ wt,
                             int K, int N, int Kpad, int Npad) {
  int idx = blockIdx.x * blockDim.x + threadIdx.x;
  if (idx >= Npad * Kpad) return;
  int k = idx % Kpad, n = idx / Kpad;
  float v = (k < K && n < N) ? w[(size_t)k * N + n] : 0.f;
  wt[idx] = (_Float16)v;
}

// ---------- area weights per (offset, b, q) ----------
__global__ void area_kernel(const float* __restrict__ coord, float* __restrict__ area) {
  int idx = blockIdx.x * blockDim.x + threadIdx.x;
  if (idx >= NOFF * BQ) return;
  int bq = idx % BQ, o = idx / BQ;
  const float* cd = coord + (size_t)bq * 18;
  int iy, ix; float qc0, qc1;
  grid_pos(cd, offv(o / 6), offv(o % 6), iy, ix, qc0, qc1);
  float r0 = (cd[0] - qc0) * 4096.f;   // even index scale = h*w
  float r1 = (cd[1] - qc1);            // odd index scale = 1
  area[idx] = fabsf(r0 * r1) + 1e-9f;
}

__global__ void tot_kernel(const float* __restrict__ area, float* __restrict__ tot) {
  int bq = blockIdx.x * blockDim.x + threadIdx.x;
  if (bq >= BQ) return;
  float s = 0.f;
#pragma unroll
  for (int o = 0; o < NOFF; ++o) s += area[(size_t)o * BQ + bq];
  tot[bq] = s;
}

// ---------- build f16 MLP input rows: q_feat + sinu, concat rel_cell ----------
__global__ void build_x_kernel(const float* __restrict__ coord,
                               const float* __restrict__ cell,
                               const float* __restrict__ feat,
                               _Float16* __restrict__ X, int co) {
  int row = blockIdx.x;               // [0, ROWS)
  int i  = row / BQ;
  int bq = row % BQ;
  int b  = bq / Qn;
  int o  = co * CH + i;
  const float* cd = coord + (size_t)bq * 18;
  int iy, ix; float qc0, qc1;
  grid_pos(cd, offv(o / 6), offv(o % 6), iy, ix, qc0, qc1);
  _Float16* xr = X + (size_t)row * KP0;
  for (int j = threadIdx.x; j < KP0; j += blockDim.x) {
    float v;
    if (j < 576) {
      // q_feat: unfolded channel cf = c*9 + s, shift s = (dy+1)*3 + (dx+1)
      int c = j / 9, s = j % 9;
      int y = iy + s / 3 - 1, x = ix + s % 3 - 1;
      float qf = 0.f;
      if ((unsigned)y < 64u && (unsigned)x < 64u)
        qf = feat[(((size_t)b * 64 + c) * 64 + y) * 64 + x];
      // sinu: pair p = j/64, component u = j%64
      int p = j >> 6, u = j & 63;
      float rel; int k; bool is_sin;
      if (u < 32) { rel = (cd[2 * p]     - qc0) * 4096.f; k = u >> 1;        is_sin = !(u & 1); }
      else        { int u2 = u - 32; rel = (cd[2 * p + 1] - qc1); k = u2 >> 1; is_sin = !(u2 & 1); }
      float ang = rel * __expf(-(float)k * LOG1E4_D16);
      v = qf + (is_sin ? __sinf(ang) : __cosf(ang));
    } else if (j == 576) {
      v = cell[(size_t)bq * 2 + 0] * 64.f;   // rel_cell = cell * [h, w]
    } else if (j == 577) {
      v = cell[(size_t)bq * 2 + 1] * 64.f;
    } else {
      v = 0.f;                                // K padding
    }
    xr[j] = (_Float16)v;
  }
}

// ---------- async staging of a 32-wide K-slab of the weight panel into LDS ----------
// LDS panel layout: [NPAD rows][32 halves], row-contiguous (64 B per row).
__device__ __forceinline__ void wait_async_all() {
#if HAS_ASYNC_LDS
#if __has_builtin(__builtin_amdgcn_s_wait_asynccnt)
  __builtin_amdgcn_s_wait_asynccnt(0);
#else
  asm volatile("s_wait_asynccnt 0x0" ::: "memory");
#endif
#endif
}

template <int NPAD>
__device__ __forceinline__ void stage_panel(const _Float16* __restrict__ Wt, int K,
                                            int k0, _Float16* lp, int tid) {
  constexpr int SEGS = NPAD * 4;          // 16-byte segments in the slab
#pragma unroll 4
  for (int seg = tid; seg < SEGS; seg += 256) {
    const int n = seg >> 2, part = seg & 3;
    const _Float16* g = Wt + (size_t)n * K + k0 + part * 8;
    _Float16* l = lp + n * 32 + part * 8;
#if HAS_ASYNC_LDS
    __builtin_amdgcn_global_load_async_to_lds_b128((g_v4i*)g, (l_v4i*)l, 0, 0);
#else
    *(v8h*)l = *(const v8h*)g;            // sync fallback: global_load + ds_store
#endif
  }
}

// ---------- WMMA GEMM: wave computes 16 x (NT*16); block shares B panel in LDS ----
// A: f16 [M][K] row-major. Wt: f16 [NT*16][K] (transposed weights).
// Grid must satisfy M/16 == gridDim.x * 8 (all waves active -> barriers safe).
template <int NT, bool RELU, bool OUT16>
__global__ void __launch_bounds__(256)
gemm_wmma_kernel(const _Float16* __restrict__ A, const _Float16* __restrict__ Wt,
                 const float* __restrict__ bias, int nbias,
                 void* __restrict__ out, int K) {
  constexpr int NPAD = NT * 16;
  constexpr int G = (NT < 4) ? NT : 4;        // B fragments loaded per group
  __shared__ __align__(64) _Float16 panel[2][NPAD * 32];
  const int tid  = threadIdx.x;
  const int lane = tid & 31;
  const int mtile = blockIdx.x * 8 + (tid >> 5);
  const int half = lane & 15;
  const bool hi  = lane >= 16;
  const _Float16* Arow = A + (size_t)(mtile * 16 + half) * K;

  v8f acc[NT] = {};
  const int S = K / 32;
  stage_panel<NPAD>(Wt, K, 0, &panel[0][0], tid);
  // Preload A fragment for slab 0 (contiguous k order matching B: lanes 0-15 hold
  // K{k0..+15}, lanes 16-31 hold K{k0+16..+31}).
  v8h alo = *(const v8h*)(Arow + (hi ? 16 : 0));
  v8h ahi = *(const v8h*)(Arow + (hi ? 24 : 8));
  for (int s = 0; s < S; ++s) {
    wait_async_all();                         // this wave's pending slab landed
    __syncthreads();                          // all waves staged; prior compute done
    if (s + 1 < S)                            // overlap next DMA with this compute
      stage_panel<NPAD>(Wt, K, (s + 1) * 32, &panel[(s + 1) & 1][0], tid);
    v16h a = __builtin_shufflevector(alo, ahi,
        0, 1, 2, 3, 4, 5, 6, 7, 8, 9, 10, 11, 12, 13, 14, 15);
    if (s + 1 < S) {                          // prefetch next A behind this compute
      const int k1 = (s + 1) * 32;
      alo = *(const v8h*)(Arow + k1 + (hi ? 16 : 0));
      ahi = *(const v8h*)(Arow + k1 + (hi ? 24 : 8));
    }
    const _Float16* pb = &panel[s & 1][0] + half * 32 + (hi ? 16 : 0);
    // Grouped B loads: issue G fragments (2G ds_load_b128), pin them above the
    // WMMAs with a sched barrier so the waitcnt pass emits staggered partial
    // dscnt waits instead of a full drain per WMMA.
#pragma unroll
    for (int g = 0; g < NT; g += G) {
      v16h bf[G];
#pragma unroll
      for (int i = 0; i < G; ++i)
        bf[i] = *(const v16h*)(pb + (g + i) * 16 * 32);
      sched_fence();
#pragma unroll
      for (int i = 0; i < G; ++i)
        acc[g + i] = __builtin_amdgcn_wmma_f32_16x16x32_f16(
            false, a, false, bf[i], (short)0, acc[g + i], false, false);
    }
  }
#pragma unroll
  for (int nt = 0; nt < NT; ++nt) {
    const int n = nt * 16 + half;
    const float bv = (n < nbias) ? bias[n] : 0.f;
#pragma unroll
    for (int r = 0; r < 8; ++r) {
      const int row = mtile * 16 + (hi ? 8 : 0) + r;   // ISA C/D layout
      float v = acc[nt][r] + bv;
      if (RELU) v = fmaxf(v, 0.f);
      if (OUT16) ((_Float16*)out)[(size_t)row * NPAD + n] = (_Float16)v;
      else       ((float*)out)[(size_t)row * NPAD + n] = v;
    }
  }
}

// ---------- blend chunk predictions into output with area/tot weights ----------
__global__ void accum_kernel(const float* __restrict__ pred,
                             const float* __restrict__ area,
                             const float* __restrict__ tot,
                             float* __restrict__ out, int co) {
  int idx = blockIdx.x * blockDim.x + threadIdx.x;
  if (idx >= BQ * NOUT) return;
  int c = idx % NOUT, bq = idx / NOUT;
  float t = tot[bq];
  float acc = 0.f;
#pragma unroll
  for (int i = 0; i < CH; ++i) {
    int o  = co * CH + i;
    int po = (o < 4) ? 3 - o : o;              // perm[[0,1,2,3]] = [3,2,1,0]
    float w = area[(size_t)po * BQ + bq] / t;
    acc += pred[((size_t)i * BQ + bq) * NP4 + c] * w;
  }
  if (co == 0) out[idx] = acc;                 // overwrites 0xAA poison deterministically
  else         out[idx] += acc;
}

extern "C" void kernel_launch(void* const* d_in, const int* in_sizes, int n_in,
                              void* d_out, int out_size, void* d_ws, size_t ws_size,
                              hipStream_t stream) {
  (void)in_sizes; (void)n_in; (void)out_size; (void)ws_size;
  const float* inp   = (const float*)d_in[0];
  const float* coord = (const float*)d_in[1];
  const float* cell  = (const float*)d_in[2];
  const float* enc_w = (const float*)d_in[3];
  const float* enc_b = (const float*)d_in[4];
  const float* w[5]  = {(const float*)d_in[5], (const float*)d_in[7],
                        (const float*)d_in[9], (const float*)d_in[11],
                        (const float*)d_in[13]};
  const float* bs[5] = {(const float*)d_in[6], (const float*)d_in[8],
                        (const float*)d_in[10], (const float*)d_in[12],
                        (const float*)d_in[14]};
  float* outp = (float*)d_out;

  uintptr_t base = (uintptr_t)d_ws;
  size_t off = 0;
  auto carve = [&](size_t bytes) -> void* {
    void* p = (void*)(base + off);
    off = (off + bytes + 255) & ~(size_t)255;
    return p;
  };
  float*    feat = (float*)   carve(sizeof(float) * Bn * 64 * 64 * 64);   // 2 MB
  _Float16* wt0  = (_Float16*)carve(sizeof(_Float16) * NMID * KP0);
  _Float16* wt1  = (_Float16*)carve(sizeof(_Float16) * NMID * NMID);
  _Float16* wt2  = (_Float16*)carve(sizeof(_Float16) * NMID * NMID);
  _Float16* wt3  = (_Float16*)carve(sizeof(_Float16) * NMID * NMID);
  _Float16* wt4  = (_Float16*)carve(sizeof(_Float16) * NP4 * NMID);
  float*    area = (float*)   carve(sizeof(float) * NOFF * BQ);
  float*    tot  = (float*)   carve(sizeof(float) * BQ);
  _Float16* X    = (_Float16*)carve(sizeof(_Float16) * (size_t)ROWS * KP0); // ~30 MB
  _Float16* actA = (_Float16*)carve(sizeof(_Float16) * (size_t)ROWS * NMID);
  _Float16* actB = (_Float16*)carve(sizeof(_Float16) * (size_t)ROWS * NMID);
  float*    pred = (float*)   carve(sizeof(float) * (size_t)ROWS * NP4);

  encoder_kernel<<<(Bn * 64 * 64 * 64 + 255) / 256, 256, 0, stream>>>(inp, enc_w, enc_b, feat);
  cvt_w_kernel<<<(NMID * KP0 + 255) / 256, 256, 0, stream>>>(w[0], wt0, K0, NMID, KP0, NMID);
  cvt_w_kernel<<<(NMID * NMID + 255) / 256, 256, 0, stream>>>(w[1], wt1, NMID, NMID, NMID, NMID);
  cvt_w_kernel<<<(NMID * NMID + 255) / 256, 256, 0, stream>>>(w[2], wt2, NMID, NMID, NMID, NMID);
  cvt_w_kernel<<<(NMID * NMID + 255) / 256, 256, 0, stream>>>(w[3], wt3, NMID, NMID, NMID, NMID);
  cvt_w_kernel<<<(NP4 * NMID + 255) / 256, 256, 0, stream>>>(w[4], wt4, NMID, NOUT, NMID, NP4);
  area_kernel<<<(NOFF * BQ + 255) / 256, 256, 0, stream>>>(coord, area);
  tot_kernel<<<(BQ + 255) / 256, 256, 0, stream>>>(area, tot);

  const int gemm_blocks = (ROWS / 16) / 8;   // 192: every wave owns exactly one mtile
  for (int co = 0; co < NCH; ++co) {
    build_x_kernel<<<ROWS, 128, 0, stream>>>(coord, cell, feat, X, co);
    gemm_wmma_kernel<16, true,  true ><<<gemm_blocks, 256, 0, stream>>>(X,    wt0, bs[0], NMID, actA, KP0);
    gemm_wmma_kernel<16, true,  true ><<<gemm_blocks, 256, 0, stream>>>(actA, wt1, bs[1], NMID, actB, NMID);
    gemm_wmma_kernel<16, true,  true ><<<gemm_blocks, 256, 0, stream>>>(actB, wt2, bs[2], NMID, actA, NMID);
    gemm_wmma_kernel<16, true,  true ><<<gemm_blocks, 256, 0, stream>>>(actA, wt3, bs[3], NMID, actB, NMID);
    gemm_wmma_kernel<2,  false, false><<<gemm_blocks, 256, 0, stream>>>(actB, wt4, bs[4], NOUT, pred, NMID);
    accum_kernel<<<(BQ * NOUT + 255) / 256, 256, 0, stream>>>(pred, area, tot, outp, co);
  }
}
